// FeedForwardNetwork_53987738911461
// MI455X (gfx1250) — matrix operand order repaired and checked by reading, not compiled, not run on hardware
//
#include <hip/hip_runtime.h>
#include <hip/hip_bf16.h>

// ---------------------------------------------------------------------------
// FFN: LayerNorm -> Linear(1024->4096)+ReLU -> Linear(4096->1024)
// x: (4,2048,1024) f32 -> out f32. 8192 tokens.
// Strategy: bf16 WMMA (v_wmma_f32_16x16x32_bf16), fp32 accumulate.
// Wave tile 32x64 (2x4 WMMA tiles) for 1.5 loads/WMMA; block tile 128x128.
// M/N/K are template constants so all tile deltas fold into the 24-bit
// global-load immediate offsets: only 2 per-lane base pointers live in VGPRs.
// ---------------------------------------------------------------------------

#define D_MODEL 1024
#define D_FF    4096
#define N_TOK   8192
#define LN_EPS  1e-5f

typedef __attribute__((ext_vector_type(16))) __bf16 v16bf;
typedef __attribute__((ext_vector_type(8)))  float  v8f;

union FragAB {
    v16bf v;
    uint4 q[2];
};

__device__ __forceinline__ __bf16 f32_to_bf16_rne(float f) {
    unsigned u = __builtin_bit_cast(unsigned, f);
    unsigned r = u + 0x7FFFu + ((u >> 16) & 1u);   // round-to-nearest-even
    unsigned short h = (unsigned short)(r >> 16);
    return __builtin_bit_cast(__bf16, h);
}

// ---------------------------------------------------------------------------
// Kernel 1: LayerNorm over last dim (1024), output bf16.
// One block (256 threads) per token row; each thread owns 4 elements (float4).
// ---------------------------------------------------------------------------
__global__ void ln_to_bf16_kernel(const float* __restrict__ x,
                                  const float* __restrict__ ln_w,
                                  const float* __restrict__ ln_b,
                                  __bf16* __restrict__ h) {
    __shared__ float s_sum[256];
    __shared__ float s_sq[256];

    const int row = blockIdx.x;
    const int tid = threadIdx.x;

    const float4* xr = reinterpret_cast<const float4*>(x + (size_t)row * D_MODEL);
    float4 v = xr[tid];

    float lsum = v.x + v.y + v.z + v.w;
    float lsq  = v.x * v.x + v.y * v.y + v.z * v.z + v.w * v.w;
    s_sum[tid] = lsum;
    s_sq[tid]  = lsq;
    __syncthreads();

    #pragma unroll
    for (int s = 128; s > 0; s >>= 1) {
        if (tid < s) {
            s_sum[tid] += s_sum[tid + s];
            s_sq[tid]  += s_sq[tid + s];
        }
        __syncthreads();
    }

    const float mu  = s_sum[0] * (1.0f / D_MODEL);
    const float var = s_sq[0] * (1.0f / D_MODEL) - mu * mu;
    const float inv = rsqrtf(var + LN_EPS);

    const int c = tid * 4;
    __bf16* ho = h + (size_t)row * D_MODEL + c;
    float g0 = ln_w[c + 0], g1 = ln_w[c + 1], g2 = ln_w[c + 2], g3 = ln_w[c + 3];
    float b0 = ln_b[c + 0], b1 = ln_b[c + 1], b2 = ln_b[c + 2], b3 = ln_b[c + 3];
    ho[0] = f32_to_bf16_rne((v.x - mu) * inv * g0 + b0);
    ho[1] = f32_to_bf16_rne((v.y - mu) * inv * g1 + b1);
    ho[2] = f32_to_bf16_rne((v.z - mu) * inv * g2 + b2);
    ho[3] = f32_to_bf16_rne((v.w - mu) * inv * g3 + b3);
}

// ---------------------------------------------------------------------------
// Kernel 2/3: tiled transpose f32 -> bf16.
// in:  R x C  (row-major f32)     out: C x R  (row-major bf16), out[c][r]=in[r][c]
// Gives the GEMM a K-contiguous, N-major B operand.
// ---------------------------------------------------------------------------
__global__ void transpose_to_bf16_kernel(const float* __restrict__ in,
                                         __bf16* __restrict__ out,
                                         int R, int C) {
    __shared__ __bf16 tile[32][33];
    const int c0 = blockIdx.x * 32;
    const int r0 = blockIdx.y * 32;
    const int tx = threadIdx.x;       // 0..31
    const int ty = threadIdx.y;       // 0..7

    #pragma unroll
    for (int i = 0; i < 4; ++i) {
        int r = r0 + ty + i * 8;
        tile[ty + i * 8][tx] = f32_to_bf16_rne(in[(size_t)r * C + c0 + tx]);
    }
    __syncthreads();
    #pragma unroll
    for (int i = 0; i < 4; ++i) {
        int c = c0 + ty + i * 8;
        out[(size_t)c * R + r0 + tx] = tile[tx][ty + i * 8];
    }
}

// ---------------------------------------------------------------------------
// GEMM: C[M x N] = A[M x K] (bf16, row-major) * Bt[N x K] (bf16, K-contiguous)
//       + bias[N], optional ReLU + bf16 output (for GEMM1) else f32 output.
//
// Block: 256 threads = 8 waves arranged 4(M) x 2(N); block tile 128 x 128.
// Each wave: 32x64 tile = 2x4 WMMA 16x16 tiles, K stepped by 32.
// Per k-step: 12 b128 loads feed 8 WMMAs (each A frag reused 4x, B frag 2x).
// All mt/nt deltas are compile-time byte offsets (<= 393216 < 2^23) and fold
// into the GLOBAL_LOAD IOFFSET field; only aPtr/bPtr live as VGPR addresses.
//
// Fragment addressing (16-bit WMMA, wave32, per ISA 7.12.2):
//   A: lane L(0..15) = row L,  K chunks [k0..k0+7] and [k0+16..k0+23]
//      lane L+16     = row L,  K chunks [k0+8..k0+15] and [k0+24..k0+31]
//   B: same K chunks as A, lane L / L+16 = col L
//   C/D: lane 0..15 -> N = lane ; VGPR r -> M = r (lanes<16) / r+8 (lanes>=16).
// ---------------------------------------------------------------------------
template <int M, int N, int K, bool RELU_BF16_OUT>
__global__ void ffn_wmma_gemm_kernel(const __bf16* __restrict__ A,
                                     const __bf16* __restrict__ Bt,
                                     const float* __restrict__ bias,
                                     void* __restrict__ Cout) {
    const int lane   = threadIdx.x & 31;
    const int waveId = threadIdx.x >> 5;

    const int waveM = waveId >> 1;           // 0..3
    const int waveN = waveId & 1;            // 0..1

    const int rowBase = blockIdx.x * 128 + waveM * 32;
    const int colBase = blockIdx.y * 128 + waveN * 64;

    const int laneLo  = lane & 15;
    const int laneHi  = lane >> 4;           // 0 or 1

    v8f acc[2][4];
    const v8f vzero = {0.f, 0.f, 0.f, 0.f, 0.f, 0.f, 0.f, 0.f};
    #pragma unroll
    for (int mt = 0; mt < 2; ++mt)
        #pragma unroll
        for (int nt = 0; nt < 4; ++nt)
            acc[mt][nt] = vzero;

    // Two per-lane base pointers; all tile deltas are immediate offsets.
    const __bf16* aPtr = A  + (size_t)(rowBase + laneLo) * K + (laneHi << 3);
    const __bf16* bPtr = Bt + (size_t)(colBase + laneLo) * K + (laneHi << 3);

    #pragma unroll 2
    for (int k0 = 0; k0 < K; k0 += 32) {
        FragAB a[2], b[4];
        // A fragments: chunks at +0 and +16 in K; row tiles at +16*K (const).
        #pragma unroll
        for (int mt = 0; mt < 2; ++mt) {
            a[mt].q[0] = *reinterpret_cast<const uint4*>(aPtr + mt * 16 * K + k0);
            a[mt].q[1] = *reinterpret_cast<const uint4*>(aPtr + mt * 16 * K + k0 + 16);
        }
        // B fragments: chunks at +0 and +16 in K (same as A); col tiles at +16*K (const).
        #pragma unroll
        for (int nt = 0; nt < 4; ++nt) {
            b[nt].q[0] = *reinterpret_cast<const uint4*>(bPtr + nt * 16 * K + k0);
            b[nt].q[1] = *reinterpret_cast<const uint4*>(bPtr + nt * 16 * K + k0 + 16);
        }

        #pragma unroll
        for (int mt = 0; mt < 2; ++mt)
            #pragma unroll
            for (int nt = 0; nt < 4; ++nt)
                acc[mt][nt] = __builtin_amdgcn_wmma_f32_16x16x32_bf16(
                    false, a[mt].v, false, b[nt].v, (short)0, acc[mt][nt],
                    false, false);
    }

    // Store: lane 0..15 -> col = colBase + nt*16 + laneLo ; rows r / r+8 by laneHi.
    const int rowOff = laneHi * 8;
    const int colL   = colBase + laneLo;
    #pragma unroll
    for (int mt = 0; mt < 2; ++mt) {
        #pragma unroll
        for (int nt = 0; nt < 4; ++nt) {
            const int col = colL + nt * 16;
            const float bv = bias[col];
            #pragma unroll
            for (int r = 0; r < 8; ++r) {
                const int row = rowBase + mt * 16 + rowOff + r;
                float val = acc[mt][nt][r] + bv;
                if (RELU_BF16_OUT) {
                    val = fmaxf(val, 0.0f);
                    reinterpret_cast<__bf16*>(Cout)[(size_t)row * N + col] =
                        f32_to_bf16_rne(val);
                } else {
                    reinterpret_cast<float*>(Cout)[(size_t)row * N + col] = val;
                }
            }
        }
    }
}

// ---------------------------------------------------------------------------
// Launch
// ---------------------------------------------------------------------------
extern "C" void kernel_launch(void* const* d_in, const int* in_sizes, int n_in,
                              void* d_out, int out_size, void* d_ws, size_t ws_size,
                              hipStream_t stream) {
    const float* x    = (const float*)d_in[0];
    const float* ln_w = (const float*)d_in[1];
    const float* ln_b = (const float*)d_in[2];
    const float* w1   = (const float*)d_in[3];
    const float* b1   = (const float*)d_in[4];
    const float* w2   = (const float*)d_in[5];
    const float* b2   = (const float*)d_in[6];
    float* out = (float*)d_out;

    // Workspace layout (bytes):
    //   h_bf16 : 8192*1024*2 = 16 MiB
    //   w1t    : 4096*1024*2 =  8 MiB   (w1t[f][d] = w1[d][f])
    //   w2t    : 1024*4096*2 =  8 MiB   (w2t[d][f] = w2[f][d])
    //   act    : 8192*4096*2 = 64 MiB   (relu(h@w1+b1) in bf16)
    char* ws = (char*)d_ws;
    __bf16* h_bf16 = (__bf16*)(ws);
    __bf16* w1t    = (__bf16*)(ws + (size_t)16 * 1024 * 1024);
    __bf16* w2t    = (__bf16*)(ws + (size_t)24 * 1024 * 1024);
    __bf16* act    = (__bf16*)(ws + (size_t)32 * 1024 * 1024);

    // 1) LayerNorm + bf16 cast.
    ln_to_bf16_kernel<<<N_TOK, 256, 0, stream>>>(x, ln_w, ln_b, h_bf16);

    // 2) w1 (1024 x 4096) -> w1t (4096 x 1024), bf16.
    {
        dim3 grid(D_FF / 32, D_MODEL / 32), block(32, 8);
        transpose_to_bf16_kernel<<<grid, block, 0, stream>>>(w1, w1t, D_MODEL, D_FF);
    }
    // 3) w2 (4096 x 1024) -> w2t (1024 x 4096), bf16.
    {
        dim3 grid(D_MODEL / 32, D_FF / 32), block(32, 8);
        transpose_to_bf16_kernel<<<grid, block, 0, stream>>>(w2, w2t, D_FF, D_MODEL);
    }

    // 4) GEMM1: act = relu(h @ w1 + b1)   [8192 x 4096], bf16 out.
    {
        dim3 grid(N_TOK / 128, D_FF / 128);
        ffn_wmma_gemm_kernel<N_TOK, D_FF, D_MODEL, true><<<grid, 256, 0, stream>>>(
            h_bf16, w1t, b1, (void*)act);
    }

    // 5) GEMM2: out = act @ w2 + b2       [8192 x 1024], f32 out.
    {
        dim3 grid(N_TOK / 128, D_MODEL / 128);
        ffn_wmma_gemm_kernel<N_TOK, D_MODEL, D_FF, false><<<grid, 256, 0, stream>>>(
            act, w2t, b2, (void*)out);
    }
}
